// BilinearModel_75118978007257
// MI455X (gfx1250) — matrix operand order repaired and checked by reading, not compiled, not run on hardware
//
#include <hip/hip_runtime.h>
#include <hip/hip_bf16.h>

#define NN 50000
#define EE 1600000
#define HH 128
#define GG 256
#define BN_EPS 1e-5f

typedef __attribute__((ext_vector_type(16))) __bf16 v16bf;
typedef __attribute__((ext_vector_type(8)))  __bf16 v8bf;
typedef __attribute__((ext_vector_type(8)))  float  v8f;

// ---------------- utility ----------------
__global__ void zero_f32(float* __restrict__ p, int n) {
    int i = blockIdx.x * blockDim.x + threadIdx.x;
    if (i < n) p[i] = 0.0f;
}

// convs_W [2,128,128] f32 (k-major)  ->  Wt [2,128,128] bf16, transposed to n-major:
// Wt[layer][n][k] = W[layer][k][n]   (makes WMMA B-fragment loads contiguous)
__global__ void wt_bf16_transpose(const float* __restrict__ in, __bf16* __restrict__ out) {
    int i = blockIdx.x * blockDim.x + threadIdx.x;   // 2*128*128
    if (i >= 2 * HH * HH) return;
    int layer = i >> 14;
    int rem   = i & 16383;
    int k = rem >> 7, n = rem & 127;
    out[(layer << 14) + n * HH + k] = (__bf16)in[i];
}

// ---------------- degree / dinv ----------------
__global__ void deg_accum(const int* __restrict__ dst, float* __restrict__ deg, int e) {
    int i = blockIdx.x * blockDim.x + threadIdx.x;
    if (i < e) atomicAdd(&deg[dst[i]], 1.0f);
}

__global__ void finalize_dinv(float* __restrict__ deg, int n) {
    int i = blockIdx.x * blockDim.x + threadIdx.x;
    if (i < n) deg[i] = rsqrtf(deg[i] + 1.0f);
}

// ---------------- layer 0 GEMM (K=7, scalar) ----------------
__global__ void gemm0_k(const float* __restrict__ x, const float* __restrict__ W,
                        float* __restrict__ T) {
    int i = blockIdx.x * blockDim.x + threadIdx.x;   // over N*H
    if (i >= NN * HH) return;
    int n = i >> 7, c = i & 127;
    float acc = 0.0f;
#pragma unroll
    for (int j = 0; j < 7; ++j) acc += x[n * 7 + j] * W[j * HH + c];
    T[i] = acc;
}

// ---------------- 128x128 GEMM via WMMA bf16 ----------------
// A:  [N,128]  bf16 row-major
// Bt: [128,128] bf16 n-major (transposed weights: Bt[n][k])
// T:  [N,128]  f32
// grid (N/16, 2), block 128 (4 waves); wave w handles col tile blockIdx.y*4+w.
__global__ void gemm128_wmma(const __bf16* __restrict__ A, const __bf16* __restrict__ Bt,
                             float* __restrict__ T) {
    const int lane = threadIdx.x & 31;
    const int wave = threadIdx.x >> 5;
    const int row0 = blockIdx.x * 16;
    const int col0 = (blockIdx.y * 4 + wave) * 16;
    const int half = lane >> 4;     // 0 or 1
    const int l16  = lane & 15;

    const __bf16* arow = A  + (size_t)(row0 + l16) * HH;   // A row for this lane (M = l16)
    const __bf16* bcol = Bt + (size_t)(col0 + l16) * HH;   // B column for this lane (N = l16)

    v8f acc = {};
#pragma unroll
    for (int ks = 0; ks < 4; ++ks) {
        const int kbase = ks * 32;
        // A 16-bit 16x32 fragment: lane holds contiguous K run [kbase+half*16, +16) (matches B)
        v16bf a = *(const v16bf*)(arow + kbase + half * 16);
        // B 16-bit 32x16 fragment: with n-major weights this is 16 contiguous bf16
        v16bf b = *(const v16bf*)(bcol + kbase + half * 16);

        acc = __builtin_amdgcn_wmma_f32_16x16x32_bf16(
            /*neg_a=*/false, a, /*neg_b=*/false, b,
            /*c_mod=*/(short)0, acc, /*reuse_a=*/false, /*reuse_b=*/false);
    }
#pragma unroll
    for (int r = 0; r < 8; ++r) {
        int row = row0 + half * 8 + r;
        T[(size_t)row * HH + (col0 + l16)] = acc[r];
    }
}

// ---------------- aggregation ----------------
// agg = dinv^2 * t + bias  (self-loop term + bias, doubles as the zero-init)
__global__ void init_agg(const float* __restrict__ t, const float* __restrict__ dinv,
                         const float* __restrict__ bias, float* __restrict__ agg) {
    int i = blockIdx.x * blockDim.x + threadIdx.x;
    if (i >= NN * HH) return;
    int n = i >> 7, c = i & 127;
    float d = dinv[n];
    agg[i] = d * d * t[i] + bias[c];
}

// one thread per (edge, 4-channel group): gather t[src], scatter-add into agg[dst]
__global__ void edge_agg(const int* __restrict__ src, const int* __restrict__ dst,
                         const float* __restrict__ dinv, const float* __restrict__ t,
                         float* __restrict__ agg) {
    int tid = blockIdx.x * blockDim.x + threadIdx.x;   // E*32 = 51.2M
    int e = tid >> 5;
    if (e >= EE) return;
    int c = (tid & 31) * 4;
    int s = src[e], d = dst[e];
    float w = dinv[s] * dinv[d];
    const float4 v = *(const float4*)(t + (size_t)s * HH + c);
    float* ap = agg + (size_t)d * HH + c;
    atomicAdd(ap + 0, w * v.x);
    atomicAdd(ap + 1, w * v.y);
    atomicAdd(ap + 2, w * v.z);
    atomicAdd(ap + 3, w * v.w);
}

// BN (eval) + ReLU; writes f32 (for pooling) and bf16 (next WMMA GEMM)
__global__ void bn_relu(const float* __restrict__ agg,
                        const float* __restrict__ gamma, const float* __restrict__ beta,
                        const float* __restrict__ mean,  const float* __restrict__ var,
                        float* __restrict__ h32, __bf16* __restrict__ hbf) {
    int i = blockIdx.x * blockDim.x + threadIdx.x;
    if (i >= NN * HH) return;
    int c = i & 127;
    float v = agg[i];
    v = gamma[c] * (v - mean[c]) * rsqrtf(var[c] + BN_EPS) + beta[c];
    v = fmaxf(v, 0.0f);
    h32[i] = v;
    hbf[i] = (__bf16)v;
}

// ---------------- global mean pool ----------------
__global__ void pool_accum(const float* __restrict__ h, const int* __restrict__ batch,
                           float* __restrict__ sums, float* __restrict__ cnt) {
    int i = blockIdx.x * blockDim.x + threadIdx.x;
    if (i >= NN * HH) return;
    int n = i >> 7, c = i & 127;
    int g = batch[n];
    atomicAdd(&sums[(size_t)g * HH + c], h[i]);
    if (c == 0) atomicAdd(&cnt[g], 1.0f);
}

__global__ void pool_div(const float* __restrict__ sums, const float* __restrict__ cnt,
                         float* __restrict__ lig) {
    int i = blockIdx.x * blockDim.x + threadIdx.x;
    if (i >= GG * HH) return;
    int g = i >> 7;
    lig[i] = sums[i] / fmaxf(cnt[g], 1.0f);
}

// ---------------- pocket MLP + bilinear collapse: M[o,i] = sum_j bilW[o,i,j]*pk[j] --------
__global__ void pocket_k(const float* __restrict__ pf,
                         const float* __restrict__ W1, const float* __restrict__ b1,
                         const float* __restrict__ W2, const float* __restrict__ b2,
                         const float* __restrict__ bilW, float* __restrict__ M) {
    __shared__ float pfs[28];
    __shared__ float a1[64];
    __shared__ float pk[64];
    int t = threadIdx.x;
    if (t < 28) pfs[t] = pf[t];
    __syncthreads();
    if (t < 64) {
        float s = b1[t];
        for (int j = 0; j < 28; ++j) s += pfs[j] * W1[j * 64 + t];
        a1[t] = fmaxf(s, 0.0f);
    }
    __syncthreads();
    if (t < 64) {
        float s = b2[t];
        for (int j = 0; j < 64; ++j) s += a1[j] * W2[j * 64 + t];
        pk[t] = s;
    }
    __syncthreads();
    for (int idx = t; idx < 64 * HH; idx += blockDim.x) {
        const float* w = bilW + (size_t)idx * 64;
        float s = 0.0f;
        for (int j = 0; j < 64; ++j) s += w[j] * pk[j];
        M[idx] = s;
    }
}

// ---------------- fused bilinear + classifier head; one block per graph ----------------
__global__ void head_k(const float* __restrict__ lig, const float* __restrict__ M,
                       const float* __restrict__ bilb,
                       const float* __restrict__ cW1, const float* __restrict__ cb1,
                       const float* __restrict__ cW2, const float* __restrict__ cb2,
                       float* __restrict__ out) {
    __shared__ float L[HH];
    __shared__ float inter[64];
    __shared__ float hid[32];
    int g = blockIdx.x, t = threadIdx.x;   // 64 threads
    L[t]      = lig[(size_t)g * HH + t];
    L[t + 64] = lig[(size_t)g * HH + 64 + t];
    __syncthreads();
    {
        float s = bilb[t];
        for (int i = 0; i < HH; ++i) s += L[i] * M[(size_t)t * HH + i];
        inter[t] = s;
    }
    __syncthreads();
    if (t < 32) {
        float s = cb1[t];
        for (int o = 0; o < 64; ++o) s += inter[o] * cW1[o * 32 + t];
        hid[t] = fmaxf(s, 0.0f);
    }
    __syncthreads();
    if (t == 0) {
        float z = cb2[0];
        for (int j = 0; j < 32; ++j) z += hid[j] * cW2[j];
        out[g] = z;
    }
}

// ---------------- host ----------------
extern "C" void kernel_launch(void* const* d_in, const int* in_sizes, int n_in,
                              void* d_out, int out_size, void* d_ws, size_t ws_size,
                              hipStream_t stream) {
    (void)in_sizes; (void)n_in; (void)out_size; (void)ws_size;
    const float* x        = (const float*)d_in[0];
    const int*   ei       = (const int*)d_in[1];      // [2,E]
    const int*   batch    = (const int*)d_in[2];
    const float* pf       = (const float*)d_in[3];
    const float* conv0_W  = (const float*)d_in[4];
    const float* conv0_b  = (const float*)d_in[5];
    const float* convs_W  = (const float*)d_in[6];    // [2,128,128]
    const float* convs_b  = (const float*)d_in[7];    // [2,128]
    const float* bn_g     = (const float*)d_in[8];
    const float* bn_b     = (const float*)d_in[9];
    const float* bn_m     = (const float*)d_in[10];
    const float* bn_v     = (const float*)d_in[11];
    const float* pW1      = (const float*)d_in[12];
    const float* pb1      = (const float*)d_in[13];
    const float* pW2      = (const float*)d_in[14];
    const float* pb2      = (const float*)d_in[15];
    const float* bilW     = (const float*)d_in[16];
    const float* bilb     = (const float*)d_in[17];
    const float* cW1      = (const float*)d_in[18];
    const float* cb1      = (const float*)d_in[19];
    const float* cW2      = (const float*)d_in[20];
    const float* cb2      = (const float*)d_in[21];
    float* out = (float*)d_out;

    const int* src = ei;
    const int* dst = ei + EE;

    // workspace carving (256B aligned)
    char* ws = (char*)d_ws;
    size_t off = 0;
    auto carve = [&](size_t bytes) -> void* {
        void* p = ws + off;
        off = (off + bytes + 255) & ~(size_t)255;
        return p;
    };
    float*  dinv = (float*)carve((size_t)NN * 4);
    float*  t    = (float*)carve((size_t)NN * HH * 4);
    float*  agg  = (float*)carve((size_t)NN * HH * 4);
    float*  h32  = (float*)carve((size_t)NN * HH * 4);
    __bf16* hbf  = (__bf16*)carve((size_t)NN * HH * 2);
    __bf16* Wbf  = (__bf16*)carve((size_t)2 * HH * HH * 2);   // transposed, n-major
    float*  sums = (float*)carve((size_t)GG * HH * 4);
    float*  cnt  = (float*)carve((size_t)GG * 4);
    float*  lig  = (float*)carve((size_t)GG * HH * 4);
    float*  M    = (float*)carve((size_t)64 * HH * 4);

    const int B256 = 256;
    const int NH_blocks = (NN * HH + B256 - 1) / B256;   // 25000

    // degree -> dinv
    zero_f32<<<(NN + B256 - 1) / B256, B256, 0, stream>>>(dinv, NN);
    deg_accum<<<(EE + B256 - 1) / B256, B256, 0, stream>>>(dst, dinv, EE);
    finalize_dinv<<<(NN + B256 - 1) / B256, B256, 0, stream>>>(dinv, NN);

    // weights -> bf16, transposed (n-major) so WMMA B-fragments are contiguous
    wt_bf16_transpose<<<(2 * HH * HH + B256 - 1) / B256, B256, 0, stream>>>(convs_W, Wbf);

    // layer-0 linear (K=7)
    gemm0_k<<<NH_blocks, B256, 0, stream>>>(x, conv0_W, t);

    for (int i = 0; i < 3; ++i) {
        const float* bias = (i == 0) ? conv0_b : (convs_b + (size_t)(i - 1) * HH);
        init_agg<<<NH_blocks, B256, 0, stream>>>(t, dinv, bias, agg);
        edge_agg<<<(EE * 32 + B256 - 1) / B256, B256, 0, stream>>>(src, dst, dinv, t, agg);
        bn_relu<<<NH_blocks, B256, 0, stream>>>(agg,
            bn_g + (size_t)i * HH, bn_b + (size_t)i * HH,
            bn_m + (size_t)i * HH, bn_v + (size_t)i * HH, h32, hbf);
        if (i < 2) {
            dim3 grid(NN / 16, 2);       // 3125 x 2, 4 waves/block -> 8 col tiles
            gemm128_wmma<<<grid, 128, 0, stream>>>(hbf, Wbf + (size_t)i * HH * HH, t);
        }
    }

    // mean pool
    zero_f32<<<(GG * HH + B256 - 1) / B256, B256, 0, stream>>>(sums, GG * HH);
    zero_f32<<<1, B256, 0, stream>>>(cnt, GG);
    pool_accum<<<NH_blocks, B256, 0, stream>>>(h32, batch, sums, cnt);
    pool_div<<<(GG * HH + B256 - 1) / B256, B256, 0, stream>>>(sums, cnt, lig);

    // pocket MLP + bilinear collapse (pk is batch-invariant)
    pocket_k<<<1, 256, 0, stream>>>(pf, pW1, pb1, pW2, pb2, bilW, M);

    // fused bilinear + classifier
    head_k<<<GG, 64, 0, stream>>>(lig, M, bilb, cW1, cb1, cW2, cb2, out);
}
